// MOELayer_61538291417528
// MI455X (gfx1250) — matrix operand order repaired and checked by reading, compiled, not run on hardware
//
#include <hip/hip_runtime.h>

typedef __attribute__((ext_vector_type(16))) __bf16 v16bf;
typedef __attribute__((ext_vector_type(8)))  __bf16 v8bf;
typedef __attribute__((ext_vector_type(8)))  float  v8f;

#define NTOK 8192
#define NEXP 8
#define MDIM 1024
#define VDIM 4096
#define CAP  2048

__device__ __forceinline__ unsigned short f2bf(float f) {
    unsigned int u = __builtin_bit_cast(unsigned int, f);
    u += 0x7fffu + ((u >> 16) & 1u);           // round-to-nearest-even
    return (unsigned short)(u >> 16);
}

// ---- CDNA5 async Global->LDS copy (ASYNCcnt path) --------------------------------------
// VDST operand = per-lane LDS byte offset (relative to wave LDS base), VADDR = 64-bit GVA.
__device__ __forceinline__ void async_load_b128(const unsigned short* g, unsigned short* l) {
    unsigned       loff = (unsigned)(unsigned long long)l;       // low 32b of flat = LDS offset
    unsigned long long ga = (unsigned long long)g;
    asm volatile("global_load_async_to_lds_b128 %0, %1, off" :: "v"(loff), "v"(ga) : "memory");
}

template <int N>
__device__ __forceinline__ void wait_async() {
#if __has_builtin(__builtin_amdgcn_s_wait_asynccnt)
    __builtin_amdgcn_s_wait_asynccnt(N);
#else
    asm volatile("s_wait_asynccnt %0" :: "i"(N));
#endif
}

// ---------------- transpose + fp32 -> bf16 convert  (in [R x C] -> out [C x R]) ----------
__global__ void __launch_bounds__(256)
k_transpose_bf16(const float* __restrict__ in, unsigned short* __restrict__ out, int R, int C) {
    __shared__ float tile[32][33];
    const int ex = blockIdx.z;
    const float* inE = in + (size_t)ex * R * C;
    unsigned short* outE = out + (size_t)ex * R * C;
    int tx = threadIdx.x, ty = threadIdx.y;       // 32 x 8
    int cBase = blockIdx.x * 32, rBase = blockIdx.y * 32;
#pragma unroll
    for (int j = 0; j < 4; ++j)
        tile[ty + 8 * j][tx] = inE[(size_t)(rBase + ty + 8 * j) * C + cBase + tx];
    __syncthreads();
#pragma unroll
    for (int j = 0; j < 4; ++j)
        outE[(size_t)(cBase + ty + 8 * j) * R + rBase + tx] =
            f2bf(tile[tx][ty + 8 * j]);
}

// ---------------- gating: logits, softmax, top-2 ---------------------------------------
__global__ void __launch_bounds__(256)
k_gating(const float* __restrict__ x, const float* __restrict__ wg,
         float* __restrict__ gates, int* __restrict__ i1, int* __restrict__ i2,
         float* __restrict__ g1, float* __restrict__ g2) {
    __shared__ float sm[NEXP][257];
    const int n = blockIdx.x, t = threadIdx.x;
    const float* xr = x + (size_t)n * MDIM;
    float acc[NEXP];
#pragma unroll
    for (int e = 0; e < NEXP; ++e) acc[e] = 0.f;
    for (int m = t; m < MDIM; m += 256) {
        float xv = xr[m];
#pragma unroll
        for (int e = 0; e < NEXP; ++e) acc[e] += xv * wg[e * MDIM + m];
    }
#pragma unroll
    for (int e = 0; e < NEXP; ++e) sm[e][t] = acc[e];
    __syncthreads();
    for (int s = 128; s > 0; s >>= 1) {
        if (t < s)
#pragma unroll
            for (int e = 0; e < NEXP; ++e) sm[e][t] += sm[e][t + s];
        __syncthreads();
    }
    if (t == 0) {
        float lg[NEXP], gt[NEXP];
#pragma unroll
        for (int e = 0; e < NEXP; ++e) lg[e] = sm[e][0];
        float mx = lg[0];
#pragma unroll
        for (int e = 1; e < NEXP; ++e) mx = fmaxf(mx, lg[e]);
        float sum = 0.f;
#pragma unroll
        for (int e = 0; e < NEXP; ++e) { gt[e] = __expf(lg[e] - mx); sum += gt[e]; }
        float inv = 1.f / sum;
#pragma unroll
        for (int e = 0; e < NEXP; ++e) { gt[e] *= inv; gates[(size_t)n * NEXP + e] = gt[e]; }
        int a1 = 0;
#pragma unroll
        for (int e = 1; e < NEXP; ++e) if (lg[e] > lg[a1]) a1 = e;
        int a2 = -1;
#pragma unroll
        for (int e = 0; e < NEXP; ++e)
            if (e != a1 && (a2 < 0 || lg[e] > lg[a2])) a2 = e;
        i1[n] = a1; i2[n] = a2; g1[n] = gt[a1]; g2[n] = gt[a2];
    }
}

// ---------------- capacity scan (one wave; lane e owns expert e) ------------------------
__global__ void k_scan(const int* __restrict__ i1, const int* __restrict__ i2,
                       const float* __restrict__ g1, const float* __restrict__ g2,
                       const float* __restrict__ gates,
                       int* __restrict__ drow1, int* __restrict__ drow2,
                       int* __restrict__ crow1, int* __restrict__ crow2,
                       float* __restrict__ s1, float* __restrict__ s2,
                       float* __restrict__ laux_out) {
    __shared__ int   scnt[NEXP];
    __shared__ float sred[NEXP];
    const int e = threadIdx.x;
    if (e < NEXP) {
        int c = 0;
        for (int n = 0; n < NTOK; ++n) c += (i1[n] == e);
        scnt[e] = c;
    }
    __syncthreads();
    if (e < NEXP) {
        const int base2 = scnt[e];
        int c1 = 0, c2 = 0, ce = 0;
        float me = 0.f;
        for (int n = 0; n < NTOK; ++n) {
            me += gates[(size_t)n * NEXP + e];
            if (i1[n] == e) {
                int loc = c1++; bool keep = loc < CAP;
                drow1[n] = keep ? e * CAP + loc : -1;
                crow1[n] = e * CAP + (keep ? loc : 0);
                s1[n]    = keep ? g1[n] : 0.f;
                ce += keep ? 1 : 0;
            }
            if (i2[n] == e) {
                int loc = base2 + c2++; bool keep = loc < CAP;
                drow2[n] = keep ? e * CAP + loc : -1;
                crow2[n] = e * CAP + (keep ? loc : 0);
                s2[n]    = keep ? g2[n] : 0.f;
            }
        }
        sred[e] = me * (float)ce;
    }
    __syncthreads();
    if (threadIdx.x == 0) {
        float s = 0.f;
        for (int q = 0; q < NEXP; ++q) s += sred[q];
        *laux_out = s * ((float)NEXP / ((float)NTOK * (float)NTOK));
    }
}

// ---------------- zero fill (uint4 grid-stride) ----------------------------------------
__global__ void k_zero(uint4* __restrict__ p, size_t n4) {
    size_t i = (size_t)blockIdx.x * blockDim.x + threadIdx.x;
    size_t stride = (size_t)gridDim.x * blockDim.x;
    uint4 z; z.x = z.y = z.z = z.w = 0u;
    for (; i < n4; i += stride) p[i] = z;
}

// ---------------- dispatch scatter (fp32 -> bf16, write-once slots) ---------------------
__global__ void __launch_bounds__(256)
k_dispatch(const float* __restrict__ x, const int* __restrict__ drow1,
           const int* __restrict__ drow2, unsigned short* __restrict__ disp) {
    const int n = blockIdx.x;
    const int r1 = drow1[n], r2 = drow2[n];
    const float* xr = x + (size_t)n * MDIM;
    for (int j = threadIdx.x; j < MDIM; j += 256) {
        unsigned short v = f2bf(xr[j]);
        if (r1 >= 0) disp[(size_t)r1 * MDIM + j] = v;
        if (r2 >= 0) disp[(size_t)r2 * MDIM + j] = v;
    }
}

// ---------------- batched WMMA GEMM: D = A[rows x K] * B^T[ncols x K] + bias ------------
// Block = 8 waves sharing one 64-row A panel; panel staged in LDS via async Global->LDS
// copies (double-buffered, 1 async b128 instruction per wave per fill). Each wave computes
// a 64x64 tile with 4x4 v_wmma_f32_16x16x32_bf16 register blocking. B (N-major) streams
// per-wave from global through L2. Grids are exact: no wave diverges around barriers.
template <bool RELU, bool OUTBF16>
__global__ void __launch_bounds__(256)
k_gemm(const unsigned short* __restrict__ A, const unsigned short* __restrict__ B,
       const float* __restrict__ bias, void* __restrict__ Out,
       int rows, int ncols, int kdim, int tilesN) {
    __shared__ unsigned short sA[2][64 * 40];   // 32-k panel, row pitch 40 elems (80B): conflict-free b128
    const int e = blockIdx.y;
    A    += (size_t)e * rows * kdim;
    B    += (size_t)e * ncols * kdim;
    bias += (size_t)e * ncols;
    const int wave = threadIdx.x >> 5;
    const int lane = threadIdx.x & 31;
    const int tileBase = blockIdx.x * 8;        // tilesN % 8 == 0 -> all 8 waves share tM
    const int tM = tileBase / tilesN;
    const int tN = tileBase % tilesN + wave;
    const int laneLo = lane & 15;
    const int laneHi = lane >> 4;

    // cooperative fill map: thread -> (row 0..63, 16B chunk 0..3)
    const int frow = threadIdx.x >> 2;
    const int fco  = (threadIdx.x & 3) * 8;     // element offset within 32-wide k chunk
    const unsigned short* gA = A + (size_t)(tM * 64 + frow) * kdim + fco;

    const int nk = kdim / 32;
    async_load_b128(gA, &sA[0][frow * 40 + fco]);           // prologue: buffer 0

    v8f acc[4][4] = {};
    const unsigned short* bBase = B + (size_t)(tN * 64 + laneLo) * kdim + laneHi * 16;

    for (int kt = 0; kt < nk; ++kt) {
        const int k = kt * 32;
        const bool more = (kt + 1) < nk;
        if (more)
            async_load_b128(gA + k + 32, &sA[(kt + 1) & 1][frow * 40 + fco]);
        if (more) wait_async<1>(); else wait_async<0>();    // current buffer complete
        __syncthreads();

        const unsigned short* sbuf = &sA[kt & 1][0];
        v16bf af[4], bb[4];
#pragma unroll
        for (int i = 0; i < 4; ++i) {
            const unsigned short* p = sbuf + (i * 16 + laneLo) * 40 + laneHi * 16;
            v8bf lo = *(const v8bf*)p;          // K = k0 .. k0+7
            v8bf hi = *(const v8bf*)(p + 8);    // K = k0+8 .. k0+15
            af[i] = __builtin_shufflevector(lo, hi, 0, 1, 2, 3, 4, 5, 6, 7,
                                            8, 9, 10, 11, 12, 13, 14, 15);
        }
#pragma unroll
        for (int i = 0; i < 4; ++i)
            bb[i] = *(const v16bf*)(bBase + (size_t)i * 16 * kdim + k);
#pragma unroll
        for (int a = 0; a < 4; ++a)
#pragma unroll
            for (int b = 0; b < 4; ++b)
                acc[a][b] = __builtin_amdgcn_wmma_f32_16x16x32_bf16(
                    false, af[a], false, bb[b], (short)0, acc[a][b], false, false);
        __syncthreads();                        // panel fully consumed before re-fill
    }

    const size_t outBase = (size_t)e * rows * ncols;
#pragma unroll
    for (int a = 0; a < 4; ++a) {
#pragma unroll
        for (int b = 0; b < 4; ++b) {
            const int col = tN * 64 + b * 16 + laneLo;
            const float bv = bias[col];
#pragma unroll
            for (int r = 0; r < 8; ++r) {
                const int row = tM * 64 + a * 16 + laneHi * 8 + r;
                float v = acc[a][b][r] + bv;
                if (RELU) v = v > 0.f ? v : 0.f;
                if (OUTBF16)
                    ((unsigned short*)Out)[outBase + (size_t)row * ncols + col] = f2bf(v);
                else
                    ((float*)Out)[outBase + (size_t)row * ncols + col] = v;
            }
        }
    }
}

// ---------------- combine: out = s1*Y[row1] + s2*Y[row2] --------------------------------
__global__ void __launch_bounds__(256)
k_combine(const float* __restrict__ Y, const int* __restrict__ crow1,
          const int* __restrict__ crow2, const float* __restrict__ s1,
          const float* __restrict__ s2, float* __restrict__ out) {
    const int n = blockIdx.x;
    const float a = s1[n], b = s2[n];
    const float4* y1 = (const float4*)(Y + (size_t)crow1[n] * MDIM);
    const float4* y2 = (const float4*)(Y + (size_t)crow2[n] * MDIM);
    float4* o = (float4*)(out + (size_t)n * MDIM);
    for (int j = threadIdx.x; j < MDIM / 4; j += 256) {
        float4 p = y1[j], q = y2[j], r;
        r.x = a * p.x + b * q.x; r.y = a * p.y + b * q.y;
        r.z = a * p.z + b * q.z; r.w = a * p.w + b * q.w;
        o[j] = r;
    }
}

extern "C" void kernel_launch(void* const* d_in, const int* in_sizes, int n_in,
                              void* d_out, int out_size, void* d_ws, size_t ws_size,
                              hipStream_t stream) {
    const float* x   = (const float*)d_in[0];   // [N, M]
    const float* wg  = (const float*)d_in[1];   // [E, M]
    const float* fc1 = (const float*)d_in[2];   // [E, M, V]
    const float* b1  = (const float*)d_in[3];   // [E, V]
    const float* fc2 = (const float*)d_in[4];   // [E, V, M]
    const float* b2  = (const float*)d_in[5];   // [E, M]
    float* out = (float*)d_out;                 // [N*M] + [1] l_aux

    // ---- workspace carve-up (256B aligned) ----
    char* ws = (char*)d_ws;
    size_t off = 0;
    auto take = [&](size_t bytes) -> char* {
        char* p = ws + off;
        off = (off + bytes + 255) & ~(size_t)255;
        return p;
    };
    float* gates = (float*)take((size_t)NTOK * NEXP * 4);
    int*   i1    = (int*)take((size_t)NTOK * 4);
    int*   i2    = (int*)take((size_t)NTOK * 4);
    float* g1    = (float*)take((size_t)NTOK * 4);
    float* g2    = (float*)take((size_t)NTOK * 4);
    int*   drow1 = (int*)take((size_t)NTOK * 4);
    int*   drow2 = (int*)take((size_t)NTOK * 4);
    int*   crow1 = (int*)take((size_t)NTOK * 4);
    int*   crow2 = (int*)take((size_t)NTOK * 4);
    float* s1    = (float*)take((size_t)NTOK * 4);
    float* s2    = (float*)take((size_t)NTOK * 4);
    unsigned short* disp = (unsigned short*)take((size_t)NEXP * CAP * MDIM * 2);   // 32 MB
    unsigned short* fc1t = (unsigned short*)take((size_t)NEXP * VDIM * MDIM * 2);  // 64 MB [E,V,M]
    unsigned short* fc2t = (unsigned short*)take((size_t)NEXP * MDIM * VDIM * 2);  // 64 MB [E,M,V]
    unsigned short* H    = (unsigned short*)take((size_t)NEXP * CAP * VDIM * 2);   // 128 MB
    float*          Y    = (float*)take((size_t)NEXP * CAP * MDIM * 4);            // 64 MB
    (void)ws_size; (void)in_sizes; (void)n_in; (void)out_size;

    // 1) convert+transpose weights to bf16 (fc1: [M,V]->[V,M], fc2: [V,M]->[M,V])
    k_transpose_bf16<<<dim3(VDIM / 32, MDIM / 32, NEXP), dim3(32, 8), 0, stream>>>(
        fc1, fc1t, MDIM, VDIM);
    k_transpose_bf16<<<dim3(MDIM / 32, VDIM / 32, NEXP), dim3(32, 8), 0, stream>>>(
        fc2, fc2t, VDIM, MDIM);

    // 2) gating
    k_gating<<<dim3(NTOK), dim3(256), 0, stream>>>(x, wg, gates, i1, i2, g1, g2);

    // 3) capacity scan + l_aux
    k_scan<<<dim3(1), dim3(32), 0, stream>>>(i1, i2, g1, g2, gates,
                                             drow1, drow2, crow1, crow2, s1, s2,
                                             out + (size_t)NTOK * MDIM);

    // 4) zero dispatch buffer, then scatter tokens (write-once per slot)
    k_zero<<<dim3(4096), dim3(256), 0, stream>>>(
        (uint4*)disp, (size_t)NEXP * CAP * MDIM * 2 / 16);
    k_dispatch<<<dim3(NTOK), dim3(256), 0, stream>>>(x, drow1, drow2, disp);

    // 5) expert GEMM1: H = relu(disp @ fc1 + b1)   [per expert: 2048x1024 @ 1024x4096]
    {
        const int tilesM = CAP / 64, tilesN = VDIM / 64;       // 32 x 64 = 2048 tiles
        const int total = tilesM * tilesN;
        k_gemm<true, true><<<dim3(total / 8, NEXP), dim3(256), 0, stream>>>(
            disp, fc1t, b1, H, CAP, VDIM, MDIM, tilesN);
    }
    // 6) expert GEMM2: Y = H @ fc2 + b2            [per expert: 2048x4096 @ 4096x1024]
    {
        const int tilesM = CAP / 64, tilesN = MDIM / 64;       // 32 x 16 = 512 tiles
        const int total = tilesM * tilesN;
        k_gemm<false, false><<<dim3(total / 8, NEXP), dim3(256), 0, stream>>>(
            H, fc2t, b2, Y, CAP, MDIM, VDIM, tilesN);
    }

    // 7) gather-combine
    k_combine<<<dim3(NTOK), dim3(256), 0, stream>>>(Y, crow1, crow2, s1, s2, out);
}
